// LinearLayer_PSLoRA_38225208934786
// MI455X (gfx1250) — matrix operand order repaired and checked by reading, not compiled, not run on hardware
//
#include <hip/hip_runtime.h>

#define D_IN   4096
#define D_OUT  4096
#define S_LEN  2048
#define B_SZ   4
#define R_DIM  32
#define KT_X   (D_IN / 32)           // 128 K-tiles for the main GEMM
#define MT_TOT ((B_SZ * S_LEN) / 16) // 512 global M-tiles (b,s flattened)
#define NT_TOT (D_OUT / 16)          // 256 N-tiles
#define TILE_BF16 1024               // one packed tile = 32 lanes * 32 bf16 (32B hi + 32B lo)

typedef __attribute__((ext_vector_type(16))) __bf16 v16bf;
typedef __attribute__((ext_vector_type(8)))  float  v8f;
typedef __attribute__((ext_vector_type(4)))  int    v4i;

// ---------------- async global -> LDS path (gfx1250), with safe fallback ----------------
#if __has_builtin(__builtin_amdgcn_global_load_async_to_lds_b128) && \
    __has_builtin(__builtin_amdgcn_s_wait_asynccnt)
#define USE_ASYNC_LDS 1
typedef __attribute__((address_space(1))) v4i gv4i;   // global (printed as __device__ by clang)
typedef __attribute__((address_space(3))) v4i lv4i;   // LDS
__device__ __forceinline__ void async_b128(const void* g, void* l) {
    __builtin_amdgcn_global_load_async_to_lds_b128((gv4i*)g, (lv4i*)l, 0, 0);
}
#else
#define USE_ASYNC_LDS 0
#endif

// ---------------- fp32 -> bf16 hi/lo split (RNE via integer ops) ----------------
__device__ __forceinline__ __bf16 f2bf(float f) {
    union { float f; unsigned u; } a; a.f = f;
    unsigned r = a.u + 0x7FFFu + ((a.u >> 16) & 1u);
    unsigned short h = (unsigned short)(r >> 16);
    return __builtin_bit_cast(__bf16, h);
}
__device__ __forceinline__ float bf2f(__bf16 b) {
    unsigned short h = __builtin_bit_cast(unsigned short, b);
    union { unsigned u; float f; } a; a.u = ((unsigned)h) << 16;
    return a.f;
}
__device__ __forceinline__ void split_hl(float v, v16bf& hi, v16bf& lo, int e) {
    __bf16 h = f2bf(v);
    hi[e] = h;
    lo[e] = f2bf(v - bf2f(h));
}

// bf16x3 split-precision accumulate: C += Ahi*Bhi + Ahi*Blo + Alo*Bhi (~fp32 accuracy)
__device__ __forceinline__ v8f wmma3(v16bf ah, v16bf al, v16bf bh, v16bf bl, v8f c) {
    c = __builtin_amdgcn_wmma_f32_16x16x32_bf16(false, ah, false, bh, (short)0, c, false, false);
    c = __builtin_amdgcn_wmma_f32_16x16x32_bf16(false, ah, false, bl, (short)0, c, false, false);
    c = __builtin_amdgcn_wmma_f32_16x16x32_bf16(false, al, false, bh, (short)0, c, false, false);
    return c;
}
// C/D 16x16 f32 layout: lane -> N=lane&15; VGPR j -> M = j + (lane>=16 ? 8 : 0)
__device__ __forceinline__ void store_acc(float* __restrict__ out, int ldo, int lane, v8f acc) {
    const int n  = lane & 15;
    const int mb = (lane & 16) ? 8 : 0;
#pragma unroll
    for (int j = 0; j < 8; ++j) out[(size_t)(mb + j) * ldo + n] = acc[j];
}

// ============ Pack pass: M x K row-major fp32 -> A-fragment tiles (hi|lo) ============
__global__ void pack_a_kernel(const float* __restrict__ src, int ld, int ktTotal,
                              __bf16* __restrict__ dst) {
    const int mt   = blockIdx.x;
    const int kt   = blockIdx.y * 8 + (threadIdx.x >> 5);
    if (kt >= ktTotal) return;
    const int lane = threadIdx.x & 31;
    const int m  = lane & 15;
    const int kb = (lane & 16) ? 16 : 0;
    const float* row = src + ((size_t)mt * 16 + m) * ld + kt * 32;
    v16bf hi, lo;
#pragma unroll
    for (int j = 0; j < 8; ++j) {
        split_hl(row[kb + 2*j    ], hi, lo, 2*j    );
        split_hl(row[kb + 2*j + 1], hi, lo, 2*j + 1);
    }
    __bf16* o = dst + (((size_t)mt * ktTotal + kt) * 32 + lane) * 32;
    *(v16bf*)(o)      = hi;
    *(v16bf*)(o + 16) = lo;
}

// ============ Pack pass: N x K row-major fp32 (W-style) -> B-fragment tiles ============
__global__ void pack_b_kernel(const float* __restrict__ src, int ld, int ktTotal, float scale,
                              __bf16* __restrict__ dst) {
    const int nt   = blockIdx.x;
    const int kt   = blockIdx.y * 8 + (threadIdx.x >> 5);
    if (kt >= ktTotal) return;
    const int lane = threadIdx.x & 31;
    const int n  = lane & 15;
    const int kb = (lane & 16) ? 16 : 0;
    const float* row = src + ((size_t)nt * 16 + n) * ld + kt * 32;
    v16bf hi, lo;
#pragma unroll
    for (int j = 0; j < 8; ++j) {
        split_hl(row[kb + 2*j    ] * scale, hi, lo, 2*j    );
        split_hl(row[kb + 2*j + 1] * scale, hi, lo, 2*j + 1);
    }
    __bf16* o = dst + (((size_t)nt * ktTotal + kt) * 32 + lane) * 32;
    *(v16bf*)(o)      = hi;
    *(v16bf*)(o + 16) = lo;
}

// ============ Pack pass: K x N row-major fp32 (lora_A natural) -> B-fragment tiles ============
__global__ void pack_loraA_kernel(const float* __restrict__ lora_A, __bf16* __restrict__ dst) {
    const int tnt  = blockIdx.x;               // lab*2 + nt  (5 labelers x 2 r-tiles)
    const int kt   = blockIdx.y * 8 + (threadIdx.x >> 5);
    if (kt >= KT_X) return;
    const int lab  = tnt >> 1;
    const int nt   = tnt & 1;
    const int lane = threadIdx.x & 31;
    const int n  = lane & 15;
    const int kb = (lane & 16) ? 16 : 0;
    const float* src = lora_A + (size_t)lab * D_IN * R_DIM;
    v16bf hi, lo;
#pragma unroll
    for (int j = 0; j < 8; ++j) {
        split_hl(src[(size_t)(kt*32 + kb + 2*j    ) * R_DIM + nt*16 + n], hi, lo, 2*j    );
        split_hl(src[(size_t)(kt*32 + kb + 2*j + 1) * R_DIM + nt*16 + n], hi, lo, 2*j + 1);
    }
    __bf16* o = dst + (((size_t)tnt * KT_X + kt) * 32 + lane) * 32;
    *(v16bf*)(o)      = hi;
    *(v16bf*)(o + 16) = lo;
}

// ===================== low[b,s,r] = x @ lora_A[lab[b]]  (fragment-native) =====================
__global__ __launch_bounds__(256)
void lora_low_kernel(const __bf16* __restrict__ A_pk, const int* __restrict__ labeler_index,
                     const __bf16* __restrict__ loraA_pk, float* __restrict__ low) {
    const int tid  = threadIdx.x;
    const int lane = tid & 31;
    const int w    = tid >> 5;
    const int mt   = blockIdx.x * 8 + w;        // global m-tile
    const int b    = mt / (S_LEN / 16);
    const int lab  = labeler_index[b];

    const __bf16* Ab = A_pk     + (size_t)mt * KT_X * TILE_BF16;
    const __bf16* B0 = loraA_pk + (size_t)(lab * 2 + 0) * KT_X * TILE_BF16;
    const __bf16* B1 = loraA_pk + (size_t)(lab * 2 + 1) * KT_X * TILE_BF16;

    v8f acc0 = {}, acc1 = {};
    for (int kt = 0; kt < KT_X; ++kt) {
        const __bf16* pa = Ab + (size_t)kt * TILE_BF16 + lane * 32;
        const __bf16* p0 = B0 + (size_t)kt * TILE_BF16 + lane * 32;
        const __bf16* p1 = B1 + (size_t)kt * TILE_BF16 + lane * 32;
        v16bf ah  = *(const v16bf*)pa,  al  = *(const v16bf*)(pa + 16);
        v16bf b0h = *(const v16bf*)p0,  b0l = *(const v16bf*)(p0 + 16);
        v16bf b1h = *(const v16bf*)p1,  b1l = *(const v16bf*)(p1 + 16);
        acc0 = wmma3(ah, al, b0h, b0l, acc0);
        acc1 = wmma3(ah, al, b1h, b1l, acc1);
    }
    float* obase = low + (size_t)mt * 16 * R_DIM;
    store_acc(obase,      R_DIM, lane, acc0);
    store_acc(obase + 16, R_DIM, lane, acc1);
}

// ---------------- shared compute step for the main GEMM ----------------
__device__ __forceinline__ void compute_step(const __bf16* lA, const __bf16* lB,
                                             int wm, int wn, int lane, v8f (&acc)[2][4]) {
    v16bf ah[2], al[2], bh[4], bl[4];
#pragma unroll
    for (int mi = 0; mi < 2; ++mi) {
        const __bf16* p = lA + (wm * 2 + mi) * TILE_BF16 + lane * 32;
        ah[mi] = *(const v16bf*)p;  al[mi] = *(const v16bf*)(p + 16);
    }
#pragma unroll
    for (int ni = 0; ni < 4; ++ni) {
        const __bf16* p = lB + (wn * 4 + ni) * TILE_BF16 + lane * 32;
        bh[ni] = *(const v16bf*)p;  bl[ni] = *(const v16bf*)(p + 16);
    }
#pragma unroll
    for (int mi = 0; mi < 2; ++mi)
#pragma unroll
        for (int ni = 0; ni < 4; ++ni)
            acc[mi][ni] = wmma3(ah[mi], al[mi], bh[ni], bl[ni], acc[mi][ni]);
}

// ======= Main GEMM: out = x@W^T + 0.5*low@loraB^T — async double-buffered LDS =======
__global__ __launch_bounds__(256)
void fused_gemm_kernel(const __bf16* __restrict__ A_pk, const __bf16* __restrict__ B_pk,
                       const __bf16* __restrict__ low_pk, const __bf16* __restrict__ loraB_pk,
                       float* __restrict__ out) {
    const int tid  = threadIdx.x;
    const int lane = tid & 31;
    const int w    = tid >> 5;
    const int wm   = w & 3;                   // 4 waves along M, 2 m-tiles each
    const int wn   = w >> 2;                  // 2 waves along N, 4 n-tiles each
    const int mtBase = blockIdx.z * (S_LEN / 16) + blockIdx.y * 8;
    const int ntBase = blockIdx.x * 8;

    v8f acc[2][4];
#pragma unroll
    for (int i = 0; i < 2; ++i)
#pragma unroll
        for (int j = 0; j < 4; ++j) acc[i][j] = {};

#if USE_ASYNC_LDS
    __shared__ __align__(16) __bf16 lA[2][8 * TILE_BF16];   // 2 x 16 KB
    __shared__ __align__(16) __bf16 lB[2][8 * TILE_BF16];   // 2 x 16 KB

    // issue kt=0 into buffer 0 (8 async b128 per wave per matrix pair)
#pragma unroll
    for (int t = 0; t < 4; ++t) {
        int c = tid + t * 256, tile = c >> 7, off = c & 127;
        async_b128(A_pk + ((size_t)(mtBase + tile) * KT_X) * TILE_BF16 + off * 8, &lA[0][c * 8]);
        async_b128(B_pk + ((size_t)(ntBase + tile) * KT_X) * TILE_BF16 + off * 8, &lB[0][c * 8]);
    }

    for (int kt = 0; kt < KT_X; ++kt) {
        const int cur = kt & 1;
        if (kt + 1 < KT_X) {
            // stream next tile into the other buffer, then wait only for the 8 older ops
#pragma unroll
            for (int t = 0; t < 4; ++t) {
                int c = tid + t * 256, tile = c >> 7, off = c & 127;
                async_b128(A_pk + ((size_t)(mtBase + tile) * KT_X + kt + 1) * TILE_BF16 + off * 8,
                           &lA[cur ^ 1][c * 8]);
                async_b128(B_pk + ((size_t)(ntBase + tile) * KT_X + kt + 1) * TILE_BF16 + off * 8,
                           &lB[cur ^ 1][c * 8]);
            }
            __builtin_amdgcn_s_wait_asynccnt(8);   // loads complete in order
        } else {
            __builtin_amdgcn_s_wait_asynccnt(0);
        }
        __syncthreads();                           // current buffer fully resident for all waves
        compute_step(lA[cur], lB[cur], wm, wn, lane, acc);
        __syncthreads();                           // buffer free before it is re-targeted
    }
#else
    __shared__ __align__(16) __bf16 lA[8 * TILE_BF16];      // 16 KB
    __shared__ __align__(16) __bf16 lB[8 * TILE_BF16];      // 16 KB
    uint4 ra[4], rb[4];
#pragma unroll
    for (int t = 0; t < 4; ++t) {
        int c = tid + t * 256, tile = c >> 7, off = c & 127;
        ra[t] = *(const uint4*)(A_pk + ((size_t)(mtBase + tile) * KT_X) * TILE_BF16 + off * 8);
        rb[t] = *(const uint4*)(B_pk + ((size_t)(ntBase + tile) * KT_X) * TILE_BF16 + off * 8);
    }
    for (int kt = 0; kt < KT_X; ++kt) {
        __syncthreads();
#pragma unroll
        for (int t = 0; t < 4; ++t) {
            int c = tid + t * 256;
            *(uint4*)(lA + c * 8) = ra[t];
            *(uint4*)(lB + c * 8) = rb[t];
        }
        __syncthreads();
        if (kt + 1 < KT_X) {
#pragma unroll
            for (int t = 0; t < 4; ++t) {
                int c = tid + t * 256, tile = c >> 7, off = c & 127;
                ra[t] = *(const uint4*)(A_pk + ((size_t)(mtBase + tile) * KT_X + kt + 1) * TILE_BF16 + off * 8);
                rb[t] = *(const uint4*)(B_pk + ((size_t)(ntBase + tile) * KT_X + kt + 1) * TILE_BF16 + off * 8);
            }
        }
        compute_step(lA, lB, wm, wn, lane, acc);
    }
#endif

    // ---- fused LoRA delta: one extra K=32 step, fragments straight from global ----
    {
        v16bf ah[2], al[2], bh[4], bl[4];
#pragma unroll
        for (int mi = 0; mi < 2; ++mi) {
            const __bf16* p = low_pk + (size_t)(mtBase + wm * 2 + mi) * TILE_BF16 + lane * 32;
            ah[mi] = *(const v16bf*)p;  al[mi] = *(const v16bf*)(p + 16);
        }
#pragma unroll
        for (int ni = 0; ni < 4; ++ni) {
            const __bf16* p = loraB_pk + (size_t)(ntBase + wn * 4 + ni) * TILE_BF16 + lane * 32;
            bh[ni] = *(const v16bf*)p;  bl[ni] = *(const v16bf*)(p + 16);
        }
#pragma unroll
        for (int mi = 0; mi < 2; ++mi)
#pragma unroll
            for (int ni = 0; ni < 4; ++ni)
                acc[mi][ni] = wmma3(ah[mi], al[mi], bh[ni], bl[ni], acc[mi][ni]);
    }

    // ---- store 128x128 block ----
#pragma unroll
    for (int mi = 0; mi < 2; ++mi)
#pragma unroll
        for (int ni = 0; ni < 4; ++ni) {
            float* obase = out + (size_t)(mtBase + wm * 2 + mi) * 16 * D_OUT
                               + (ntBase + wn * 4 + ni) * 16;
            store_acc(obase, D_OUT, lane, acc[mi][ni]);
        }
}

extern "C" void kernel_launch(void* const* d_in, const int* in_sizes, int n_in,
                              void* d_out, int out_size, void* d_ws, size_t ws_size,
                              hipStream_t stream) {
    (void)in_sizes; (void)n_in; (void)out_size; (void)ws_size;
    const float* x        = (const float*)d_in[0];
    const int*   lab      = (const int*)  d_in[1];
    const float* W        = (const float*)d_in[2];
    const float* lora_A   = (const float*)d_in[3];
    const float* lora_B_w = (const float*)d_in[4];
    float* out = (float*)d_out;

    // workspace layout (bytes)
    char* ws = (char*)d_ws;
    __bf16* A_pk     = (__bf16*)(ws);                                  // 512*128*2048 = 134 MB
    __bf16* B_pk     = (__bf16*)(ws + 134217728);                      // 256*128*2048 =  67 MB
    __bf16* loraA_pk = (__bf16*)(ws + 134217728 + 67108864);           // 10*128*2048  = 2.5 MB
    __bf16* loraB_pk = (__bf16*)(ws + 134217728 + 67108864 + 2621440); // 256*2048     = 0.5 MB
    float*  low_f32  = (float*) (ws + 134217728 + 67108864 + 2621440 + 524288);   // 1 MB
    __bf16* low_pk   = (__bf16*)(ws + 134217728 + 67108864 + 2621440 + 524288 + 1048576); // 1 MB

    // 1) one-time fp32 -> bf16 hi/lo fragment packing (streaming, ~200 MB total)
    pack_a_kernel<<<dim3(MT_TOT, KT_X / 8), 256, 0, stream>>>(x, D_IN, KT_X, A_pk);
    pack_b_kernel<<<dim3(NT_TOT, KT_X / 8), 256, 0, stream>>>(W, D_IN, KT_X, 1.0f, B_pk);
    pack_loraA_kernel<<<dim3(10, KT_X / 8), 256, 0, stream>>>(lora_A, loraA_pk);
    pack_b_kernel<<<dim3(NT_TOT, 1), 32, 0, stream>>>(lora_B_w, R_DIM, 1, 0.5f, loraB_pk);

    // 2) LoRA low-rank projection (fragment-native WMMA)
    lora_low_kernel<<<dim3(MT_TOT / 8), 256, 0, stream>>>(A_pk, lab, loraA_pk, low_f32);
    pack_a_kernel<<<dim3(MT_TOT, 1), 32, 0, stream>>>(low_f32, R_DIM, 1, low_pk);

    // 3) main GEMM with fused LoRA epilogue (async double-buffered if available)
    fused_gemm_kernel<<<dim3(D_OUT / 128, S_LEN / 128, B_SZ), 256, 0, stream>>>(
        A_pk, B_pk, low_pk, loraB_pk, out);
}
